// WaveNet_45595372815178
// MI455X (gfx1250) — matrix operand order repaired and checked by reading, not compiled, not run on hardware
//
#include <hip/hip_runtime.h>
#include <hip/hip_bf16.h>

typedef __attribute__((ext_vector_type(16))) _Float16 v16h;
typedef __attribute__((ext_vector_type(8)))  _Float16 v8h;
typedef __attribute__((ext_vector_type(8)))  float    v8f;

#define WMMA16(A,B,C) __builtin_amdgcn_wmma_f32_16x16x32_f16(false,(A),false,(B),(short)0,(C),false,false)

#define NBLK 30
#define RC   128
#define BSZ  2
#define CPAD 96     // cond channels padded 80 -> 96 (3 K-blocks of 32)
#define TT   64     // timesteps per workgroup in layer kernel (4 subtiles of 16)
#define INV_SQRT2 0.70710678118f

// ---------------------------------------------------------------------------
// Weight packing: f32 [O][I](stride) -> f16 A-fragments, contiguous-k (not ISA table order) 16-bit A 16x32
// layout: lane L holds row m = mb*16 + (L&15); half idx "idx" (0..15) maps to
// VGPR j=idx/2, elem e=idx&1 with k = kb*32 + (L/16)*16 + idx
// ---------------------------------------------------------------------------
__global__ void __launch_bounds__(512) pack_w_kernel(
    const float* __restrict__ src, _Float16* __restrict__ dst,
    int I, int KB, int i_stride, int elem_off,
    long src_layer_stride, long dst_layer_stride)
{
  int tid  = threadIdx.x;
  int lane = tid >> 4;          // 0..31
  int idx  = tid & 15;          // 0..15 halves per lane
  int frag = blockIdx.x;        // mb*KB + kb
  int layer= blockIdx.y;
  int mb = frag / KB;
  int kb = frag - mb * KB;
  int j = idx >> 1, e = idx & 1;
  int k = kb*32 + ((lane>>4)<<4) + idx;
  int m = (mb<<4) + (lane & 15);
  float v = (k < I) ? src[layer*src_layer_stride + ((long)m*I + k)*i_stride + elem_off] : 0.0f;
  dst[layer*dst_layer_stride + (long)frag*512 + lane*16 + idx] = (_Float16)v;
}

// ---------------------------------------------------------------------------
// Transpose+convert: src f32 [b][Cs][T] -> dst f16 [b][t][Cd] (zero pad c>=Cs)
// ---------------------------------------------------------------------------
__global__ void __launch_bounds__(256) transpose_cvt_kernel(
    const float* __restrict__ src, _Float16* __restrict__ dst,
    int Cs, int Cd, int T, long N)
{
  long i = (long)blockIdx.x * blockDim.x + threadIdx.x;
  if (i >= N) return;
  int  c  = (int)(i % Cd);
  long bt = i / Cd;
  int  t  = (int)(bt % T);
  int  b  = (int)(bt / T);
  float v = (c < Cs) ? src[((long)b*Cs + c)*T + t] : 0.0f;
  dst[i] = (_Float16)v;
}

// ---------------------------------------------------------------------------
// Input 1x1: h0 = input_w @ x + b  (M=128, K=256); also zero skip accumulator
// ---------------------------------------------------------------------------
__global__ void __launch_bounds__(256) input_kernel(
    const _Float16* __restrict__ xT, const _Float16* __restrict__ win,
    const float* __restrict__ bin, _Float16* __restrict__ h_out,
    float* __restrict__ skip, int T)
{
  __shared__ alignas(32) _Float16 s_x[16*256];
  const int tid  = threadIdx.x;
  const int lane = tid & 31;
  const int wave = tid >> 5;
  const int t0   = blockIdx.x * 16;
  const int b    = blockIdx.y;
  {
    int tt = tid >> 4, part = tid & 15;   // 16 rows x 16 parts of 16 halves
    ((v16h*)s_x)[tt*16 + part] =
        *((const v16h*)(xT + (long)b*T*256 + (long)(t0+tt)*256 + part*16));
  }
  __syncthreads();
  const int n = lane & 15, half = lane >> 4;
  v8f acc = {};
  #pragma unroll
  for (int kb = 0; kb < 8; ++kb) {
    v16h bfrag = *((const v16h*)(s_x + n*256 + kb*32 + half*16));
    v16h afrag = *((const v16h*)(win + (wave*8 + kb)*512 + lane*16));
    acc = WMMA16(afrag, bfrag, acc);
  }
  const int crow = wave*16 + half*8;
  v8f bb = *((const v8f*)(bin + crow));
  long outoff = (long)b*T*128 + (long)(t0+n)*128 + crow;
  v8h hv; v8f zs = {};
  #pragma unroll
  for (int r = 0; r < 8; ++r) hv[r] = (_Float16)(acc[r] + bb[r]);
  *((v8h*)(h_out + outoff)) = hv;
  *((v8f*)(skip + outoff))  = zs;
}

// ---------------------------------------------------------------------------
// One WaveNet residual block, 64-timestep tile, K-outer / N-inner so every
// weight A-fragment is loaded once and reused across 4 WMMA N-subtiles.
// ---------------------------------------------------------------------------
__global__ void __launch_bounds__(256) layer_kernel(
    const _Float16* __restrict__ h_in, _Float16* __restrict__ h_out,
    const _Float16* __restrict__ condT, float* __restrict__ skip,
    const _Float16* __restrict__ wdil0, const _Float16* __restrict__ wdil1,
    const _Float16* __restrict__ wgate, const _Float16* __restrict__ wcond,
    const _Float16* __restrict__ wcondg, const _Float16* __restrict__ wres,
    const _Float16* __restrict__ wskip,
    const float* __restrict__ bdil,  const float* __restrict__ bgate,
    const float* __restrict__ bcond, const float* __restrict__ bcondg,
    const float* __restrict__ bres,  const float* __restrict__ bskip,
    int dil, int T)
{
  __shared__ alignas(32) _Float16 s_hc[TT*128];   // h(t)    tile [t][c]
  __shared__ alignas(32) _Float16 s_hd[TT*128];   // h(t-d)  tile
  __shared__ alignas(32) _Float16 s_cd[TT*CPAD];  // cond    tile
  __shared__ alignas(32) _Float16 s_out[TT*128];  // gated out tile

  const int tid  = threadIdx.x;
  const int lane = tid & 31;
  const int wave = tid >> 5;
  const int t0   = blockIdx.x * TT;
  const int b    = blockIdx.y;
  const long hbase = (long)b*T*128;
  const long cbase = (long)b*T*CPAD;

  // ---- cooperative loads into LDS (4 passes of 16 rows) ----
  {
    int tt0 = tid >> 4, part = tid & 15;            // parts of 8 halves
    #pragma unroll
    for (int p = 0; p < 4; ++p) {
      int tt = tt0 + p*16;
      ((v8h*)s_hc)[tt*16 + part] =
          *((const v8h*)(h_in + hbase + (long)(t0+tt)*128 + part*8));
      int td = t0 + tt - dil;
      v8h z = {};
      ((v8h*)s_hd)[tt*16 + part] = (td >= 0)
          ? *((const v8h*)(h_in + hbase + (long)td*128 + part*8)) : z;
      if (part < 12)
        ((v8h*)s_cd)[tt*12 + part] =
            *((const v8h*)(condT + cbase + (long)(t0+tt)*CPAD + part*8));
    }
  }
  __syncthreads();

  const int n = lane & 15, half = lane >> 4;
  const int crow = wave*16 + half*8;

  // ---- a = dil0@h(t-d) + dil1@h(t) + cond@c ; g = gate@h + condg@c ----
  v8f acc_a[4], acc_g[4];
  #pragma unroll
  for (int nt = 0; nt < 4; ++nt) { v8f z = {}; acc_a[nt] = z; acc_g[nt] = z; }

  #pragma unroll
  for (int kb = 0; kb < 4; ++kb) {
    v16h a1 = *((const v16h*)(wdil1 + (wave*4 + kb)*512 + lane*16));
    v16h ag = *((const v16h*)(wgate + (wave*4 + kb)*512 + lane*16));
    v16h a0 = *((const v16h*)(wdil0 + (wave*4 + kb)*512 + lane*16));
    #pragma unroll
    for (int nt = 0; nt < 4; ++nt) {
      v16h bh = *((const v16h*)(s_hc + (nt*16 + n)*128 + kb*32 + half*16));
      acc_a[nt] = WMMA16(a1, bh, acc_a[nt]);
      acc_g[nt] = WMMA16(ag, bh, acc_g[nt]);
      v16h bd = *((const v16h*)(s_hd + (nt*16 + n)*128 + kb*32 + half*16));
      acc_a[nt] = WMMA16(a0, bd, acc_a[nt]);
    }
  }
  #pragma unroll
  for (int kb = 0; kb < 3; ++kb) {
    v16h ac = *((const v16h*)(wcond  + (wave*3 + kb)*512 + lane*16));
    v16h ah = *((const v16h*)(wcondg + (wave*3 + kb)*512 + lane*16));
    #pragma unroll
    for (int nt = 0; nt < 4; ++nt) {
      v16h bc = *((const v16h*)(s_cd + (nt*16 + n)*CPAD + kb*32 + half*16));
      acc_a[nt] = WMMA16(ac, bc, acc_a[nt]);
      acc_g[nt] = WMMA16(ah, bc, acc_g[nt]);
    }
  }

  v8f ba = *((const v8f*)(bdil  + crow));
  v8f b2 = *((const v8f*)(bcond + crow));
  v8f bg = *((const v8f*)(bgate + crow));
  v8f b4 = *((const v8f*)(bcondg+ crow));

  // ---- out = tanh(a) * sigmoid(g) ; exchange as f16 via LDS ----
  #pragma unroll
  for (int nt = 0; nt < 4; ++nt) {
    v8h oh;
    #pragma unroll
    for (int r = 0; r < 8; ++r) {
      float av = acc_a[nt][r] + ba[r] + b2[r];
      float gv = acc_g[nt][r] + bg[r] + b4[r];
      float o  = tanhf(av) * (1.0f / (1.0f + __expf(-gv)));
      oh[r] = (_Float16)o;
    }
    *((v8h*)(s_out + (nt*16 + n)*128 + crow)) = oh;
  }
  __syncthreads();

  // ---- res = res_w@out ; sk = skip_w@out ----
  v8f acc_r[4], acc_s[4];
  #pragma unroll
  for (int nt = 0; nt < 4; ++nt) { v8f z = {}; acc_r[nt] = z; acc_s[nt] = z; }

  #pragma unroll
  for (int kb = 0; kb < 4; ++kb) {
    v16h ar = *((const v16h*)(wres  + (wave*4 + kb)*512 + lane*16));
    v16h as = *((const v16h*)(wskip + (wave*4 + kb)*512 + lane*16));
    #pragma unroll
    for (int nt = 0; nt < 4; ++nt) {
      v16h bo = *((const v16h*)(s_out + (nt*16 + n)*128 + kb*32 + half*16));
      acc_r[nt] = WMMA16(ar, bo, acc_r[nt]);
      acc_s[nt] = WMMA16(as, bo, acc_s[nt]);
    }
  }
  v8f br = *((const v8f*)(bres  + crow));
  v8f bs = *((const v8f*)(bskip + crow));

  #pragma unroll
  for (int nt = 0; nt < 4; ++nt) {
    v8h hcur = *((const v8h*)(s_hc + (nt*16 + n)*128 + crow));
    long outoff = hbase + (long)(t0 + nt*16 + n)*128 + crow;
    v8f sold = *((const v8f*)(skip + outoff));
    v8h hnew; v8f snew;
    #pragma unroll
    for (int r = 0; r < 8; ++r) {
      float hv = (float)hcur[r];
      hnew[r] = (_Float16)((hv + acc_r[nt][r] + br[r]) * INV_SQRT2);
      snew[r] = sold[r] + acc_s[nt][r] + bs[r];
    }
    *((v8h*)(h_out + outoff)) = hnew;
    *((v8f*)(skip  + outoff)) = snew;
  }
}

// ---------------------------------------------------------------------------
// Head: y = relu(out1_w@skip + b1); out = out2_w@y + b2 -> d_out [b][256][t]
// ---------------------------------------------------------------------------
__global__ void __launch_bounds__(256) head_kernel(
    const float* __restrict__ skip,
    const _Float16* __restrict__ w1, const float* __restrict__ b1,
    const _Float16* __restrict__ w2, const float* __restrict__ b2,
    float* __restrict__ out, int T)
{
  __shared__ alignas(32) _Float16 s_s[16*128];
  __shared__ alignas(32) _Float16 s_y[16*128];
  const int tid  = threadIdx.x;
  const int lane = tid & 31;
  const int wave = tid >> 5;
  const int t0   = blockIdx.x * 16;
  const int b    = blockIdx.y;
  {
    int tt = tid >> 4, part = tid & 15;
    v8f sv = *((const v8f*)(skip + (long)b*T*128 + (long)(t0+tt)*128 + part*8));
    v8h sh;
    #pragma unroll
    for (int r = 0; r < 8; ++r) sh[r] = (_Float16)sv[r];
    ((v8h*)s_s)[tt*16 + part] = sh;
  }
  __syncthreads();
  const int n = lane & 15, half = lane >> 4;
  const int crow = wave*16 + half*8;

  v8f acc = {};
  #pragma unroll
  for (int kb = 0; kb < 4; ++kb) {
    v16h bf = *((const v16h*)(s_s + n*128 + kb*32 + half*16));
    v16h af = *((const v16h*)(w1 + (wave*4 + kb)*512 + lane*16));
    acc = WMMA16(af, bf, acc);
  }
  v8f bb = *((const v8f*)(b1 + crow));
  v8h yh;
  #pragma unroll
  for (int r = 0; r < 8; ++r) {
    float v = acc[r] + bb[r];
    yh[r] = (_Float16)(v > 0.0f ? v : 0.0f);
  }
  *((v8h*)(s_y + n*128 + crow)) = yh;
  __syncthreads();

  #pragma unroll
  for (int mh = 0; mh < 2; ++mh) {        // 256 output rows = 2 blocks/wave
    int mb = wave + mh*8;
    v8f a2 = {};
    #pragma unroll
    for (int kb = 0; kb < 4; ++kb) {
      v16h bf = *((const v16h*)(s_y + n*128 + kb*32 + half*16));
      v16h af = *((const v16h*)(w2 + (mb*4 + kb)*512 + lane*16));
      a2 = WMMA16(af, bf, a2);
    }
    int orow = mb*16 + half*8;
    v8f bo = *((const v8f*)(b2 + orow));
    #pragma unroll
    for (int r = 0; r < 8; ++r)
      out[((long)b*256 + orow + r)*T + t0 + n] = a2[r] + bo[r];
  }
}

// ---------------------------------------------------------------------------
extern "C" void kernel_launch(void* const* d_in, const int* in_sizes, int n_in,
                              void* d_out, int out_size, void* d_ws, size_t ws_size,
                              hipStream_t stream) {
  const float* d_x      = (const float*)d_in[0];
  const float* d_cond   = (const float*)d_in[1];
  const float* d_in_w   = (const float*)d_in[2];
  const float* d_in_b   = (const float*)d_in[3];
  const float* d_dil_w  = (const float*)d_in[4];
  const float* d_dil_b  = (const float*)d_in[5];
  const float* d_gate_w = (const float*)d_in[6];
  const float* d_gate_b = (const float*)d_in[7];
  const float* d_cond_w = (const float*)d_in[8];
  const float* d_cond_b = (const float*)d_in[9];
  const float* d_cndg_w = (const float*)d_in[10];
  const float* d_cndg_b = (const float*)d_in[11];
  const float* d_res_w  = (const float*)d_in[12];
  const float* d_res_b  = (const float*)d_in[13];
  const float* d_skip_w = (const float*)d_in[14];
  const float* d_skip_b = (const float*)d_in[15];
  const float* d_o1_w   = (const float*)d_in[16];
  const float* d_o1_b   = (const float*)d_in[17];
  const float* d_o2_w   = (const float*)d_in[18];
  const float* d_o2_b   = (const float*)d_in[19];
  float* out = (float*)d_out;

  const int T = in_sizes[0] / (BSZ * 256);
  const int ntiles16 = T / 16;
  const int ntiles64 = T / TT;

  // ---- workspace carve-up (256B aligned) ----
  char* ws = (char*)d_ws;
  size_t off = 0;
  #define CARVE(nbytes) ({ size_t o_ = off; off = (off + (size_t)(nbytes) + 255) & ~(size_t)255; o_; })
  const size_t FR = 512;  // halves per fragment
  _Float16* wp_dil0 = (_Float16*)(ws + CARVE(NBLK*8*4*FR*2));
  _Float16* wp_dil1 = (_Float16*)(ws + CARVE(NBLK*8*4*FR*2));
  _Float16* wp_gate = (_Float16*)(ws + CARVE(NBLK*8*4*FR*2));
  _Float16* wp_cond = (_Float16*)(ws + CARVE(NBLK*8*3*FR*2));
  _Float16* wp_cndg = (_Float16*)(ws + CARVE(NBLK*8*3*FR*2));
  _Float16* wp_res  = (_Float16*)(ws + CARVE(NBLK*8*4*FR*2));
  _Float16* wp_skip = (_Float16*)(ws + CARVE(NBLK*8*4*FR*2));
  _Float16* wp_in   = (_Float16*)(ws + CARVE(8*8*FR*2));
  _Float16* wp_o1   = (_Float16*)(ws + CARVE(8*4*FR*2));
  _Float16* wp_o2   = (_Float16*)(ws + CARVE(16*4*FR*2));
  _Float16* xT      = (_Float16*)(ws + CARVE((size_t)BSZ*T*256*2));
  _Float16* condT   = (_Float16*)(ws + CARVE((size_t)BSZ*T*CPAD*2));
  _Float16* hA      = (_Float16*)(ws + CARVE((size_t)BSZ*T*128*2));
  _Float16* hB      = (_Float16*)(ws + CARVE((size_t)BSZ*T*128*2));
  float*    skipb   = (float*)   (ws + CARVE((size_t)BSZ*T*128*4));
  #undef CARVE
  (void)ws_size; (void)n_in; (void)out_size;

  // ---- pack weights into WMMA A-fragment layout ----
  pack_w_kernel<<<dim3(8*4, NBLK), 512, 0, stream>>>(d_dil_w,  wp_dil0, 128, 4, 2, 0, 128L*128*2, 8L*4*FR);
  pack_w_kernel<<<dim3(8*4, NBLK), 512, 0, stream>>>(d_dil_w,  wp_dil1, 128, 4, 2, 1, 128L*128*2, 8L*4*FR);
  pack_w_kernel<<<dim3(8*4, NBLK), 512, 0, stream>>>(d_gate_w, wp_gate, 128, 4, 1, 0, 128L*128,   8L*4*FR);
  pack_w_kernel<<<dim3(8*3, NBLK), 512, 0, stream>>>(d_cond_w, wp_cond,  80, 3, 1, 0, 128L*80,    8L*3*FR);
  pack_w_kernel<<<dim3(8*3, NBLK), 512, 0, stream>>>(d_cndg_w, wp_cndg,  80, 3, 1, 0, 128L*80,    8L*3*FR);
  pack_w_kernel<<<dim3(8*4, NBLK), 512, 0, stream>>>(d_res_w,  wp_res,  128, 4, 1, 0, 128L*128,   8L*4*FR);
  pack_w_kernel<<<dim3(8*4, NBLK), 512, 0, stream>>>(d_skip_w, wp_skip, 128, 4, 1, 0, 128L*128,   8L*4*FR);
  pack_w_kernel<<<dim3(8*8, 1),    512, 0, stream>>>(d_in_w,   wp_in,   256, 8, 1, 0, 0, 0);
  pack_w_kernel<<<dim3(8*4, 1),    512, 0, stream>>>(d_o1_w,   wp_o1,   128, 4, 1, 0, 0, 0);
  pack_w_kernel<<<dim3(16*4, 1),   512, 0, stream>>>(d_o2_w,   wp_o2,   128, 4, 1, 0, 0, 0);

  // ---- transpose/convert activations to [b][t][c] f16 ----
  {
    long Nx = (long)BSZ*T*256;
    transpose_cvt_kernel<<<(unsigned)((Nx+255)/256), 256, 0, stream>>>(d_x, xT, 256, 256, T, Nx);
    long Nc = (long)BSZ*T*CPAD;
    transpose_cvt_kernel<<<(unsigned)((Nc+255)/256), 256, 0, stream>>>(d_cond, condT, 80, CPAD, T, Nc);
  }

  // ---- input 1x1 (also zeroes skip accumulator) ----
  input_kernel<<<dim3(ntiles16, BSZ), 256, 0, stream>>>(xT, wp_in, d_in_b, hA, skipb, T);

  // ---- 30 residual blocks, ping-pong h ----
  for (int i = 0; i < NBLK; ++i) {
    const _Float16* hin = (i & 1) ? hB : hA;
    _Float16*       hout= (i & 1) ? hA : hB;
    int d = 1 << (i % 10);
    layer_kernel<<<dim3(ntiles64, BSZ), 256, 0, stream>>>(
        hin, hout, condT, skipb,
        wp_dil0 + (long)i*8*4*FR, wp_dil1 + (long)i*8*4*FR,
        wp_gate + (long)i*8*4*FR, wp_cond + (long)i*8*3*FR,
        wp_cndg + (long)i*8*3*FR, wp_res  + (long)i*8*4*FR,
        wp_skip + (long)i*8*4*FR,
        d_dil_b  + i*128, d_gate_b + i*128,
        d_cond_b + i*128, d_cndg_b + i*128,
        d_res_b  + i*128, d_skip_b + i*128,
        d, T);
  }

  // ---- output head ----
  head_kernel<<<dim3(ntiles16, BSZ), 256, 0, stream>>>(skipb, wp_o1, d_o1_b, wp_o2, d_o2_b, out, T);
}
